// Packetbnn_37391985279512
// MI455X (gfx1250) — compile-verified, matrix operand order checked by reading, not run on hardware
//
#include <hip/hip_runtime.h>

// CDNA5 WMMA types
typedef __attribute__((ext_vector_type(16))) _Float16 v16h;
typedef __attribute__((ext_vector_type(8)))  float    v8f;

// (sign(x-0.5)+1)/2 -> {0, 0.5, 1}; handles the x==0.5 -> 0.5 case exactly.
__device__ __forceinline__ float binf(float x) {
  return x > 0.5f ? 1.0f : (x < 0.5f ? 0.0f : 0.5f);
}

__device__ __forceinline__ unsigned pack2(_Float16 a, _Float16 b) {
  union { _Float16 h[2]; unsigned u; } t;
  t.h[0] = a; t.h[1] = b; return t.u;
}
__device__ __forceinline__ _Float16 half_lo(unsigned u) {
  union { unsigned u; _Float16 h[2]; } t; t.u = u; return t.h[0];
}
__device__ __forceinline__ _Float16 half_hi(unsigned u) {
  union { unsigned u; _Float16 h[2]; } t; t.u = u; return t.h[1];
}

// One wave (32 lanes) handles 32 samples per pass via two WMMAs.
// A = binarized conv weights (16x32 f16, rows 0..4 / K 0..4 live, same for all waves).
// B = 16 binarized samples as columns (K 0..4 live).
// D: lane n (0..15), VGPR r -> conv_out[row r][sample n]  => stage 2 is per-lane.
__global__ __launch_bounds__(256) void bnn_wmma_kernel(
    const float* __restrict__ x, const float* __restrict__ wc,
    const float* __restrict__ wl, float* __restrict__ out, int n) {
  const int tid  = (int)(blockIdx.x * blockDim.x + threadIdx.x);
  const int lane = (int)(threadIdx.x & 31u);
  const int s0   = tid & ~31;   // first of this wave's 32 samples

  // ---------- A matrix: binarized conv weight (value-masked, EXEC stays full)
  v16h A;
  #pragma unroll
  for (int i = 0; i < 16; ++i) A[i] = (_Float16)0.0f;
  {
    const int m = (lane < 5) ? lane : 0;     // clamp index, mask value below
    #pragma unroll
    for (int k = 0; k < 5; ++k) {
      float w  = wc[m * 5 + k];
      float bw = (lane < 5) ? binf(w) : 0.0f;
      A[k] = (_Float16)bw;
    }
  }

  // ---------- stream this lane's sample: 5 f32, stride 5 (wave covers 640B contiguous)
  int s  = s0 + lane;
  int sc = (s < n) ? s : (n - 1);
  const float* xs = x + (long long)sc * 5;
  float x0 = __builtin_nontemporal_load(xs + 0);
  float x1 = __builtin_nontemporal_load(xs + 1);
  float x2 = __builtin_nontemporal_load(xs + 2);
  float x3 = __builtin_nontemporal_load(xs + 3);
  float x4 = __builtin_nontemporal_load(xs + 4);
  _Float16 h0 = (_Float16)binf(x0);
  _Float16 h1 = (_Float16)binf(x1);
  _Float16 h2 = (_Float16)binf(x2);
  _Float16 h3 = (_Float16)binf(x3);
  _Float16 h4 = (_Float16)binf(x4);

  // bring lanes 16..31 data to lanes 0..15 for the second WMMA
  unsigned p0 = pack2(h0, h1);
  unsigned p1 = pack2(h2, h3);
  unsigned p2 = pack2(h4, (_Float16)0.0f);
  unsigned q0 = __shfl_down(p0, 16, 32);
  unsigned q1 = __shfl_down(p1, 16, 32);
  unsigned q2 = __shfl_down(p2, 16, 32);

  const bool act = (lane < 16);
  const _Float16 z = (_Float16)0.0f;
  v16h B1, B2;
  #pragma unroll
  for (int i = 0; i < 16; ++i) { B1[i] = z; B2[i] = z; }
  B1[0] = act ? h0 : z;  B1[1] = act ? h1 : z;
  B1[2] = act ? h2 : z;  B1[3] = act ? h3 : z;
  B1[4] = act ? h4 : z;
  B2[0] = act ? half_lo(q0) : z;  B2[1] = act ? half_hi(q0) : z;
  B2[2] = act ? half_lo(q1) : z;  B2[3] = act ? half_hi(q1) : z;
  B2[4] = act ? half_lo(q2) : z;

  // ---------- conv stage on the matrix core (exact: operands in {0,0.5,1})
  v8f c1 = {};
  v8f c2 = {};
  c1 = __builtin_amdgcn_wmma_f32_16x16x32_f16(false, A, false, B1,
                                              (short)0, c1, false, false);
  c2 = __builtin_amdgcn_wmma_f32_16x16x32_f16(false, A, false, B2,
                                              (short)0, c2, false, false);

  // ---------- stage 2: binarize hidden, dot with binarized linear weight (per-lane)
  float w0 = binf(wl[0]), w1 = binf(wl[1]), w2 = binf(wl[2]),
        w3 = binf(wl[3]), w4 = binf(wl[4]);
  float o1 = binf(c1[0]) * w0 + binf(c1[1]) * w1 + binf(c1[2]) * w2 +
             binf(c1[3]) * w3 + binf(c1[4]) * w4;
  float o2 = binf(c2[0]) * w0 + binf(c2[1]) * w1 + binf(c2[2]) * w2 +
             binf(c2[3]) * w3 + binf(c2[4]) * w4;

  if (act) {
    int sA = s0 + lane;        // column n of B1 = sample s0+n
    int sB = s0 + 16 + lane;   // column n of B2 = sample s0+16+n
    if (sA < n) __builtin_nontemporal_store(o1, out + sA);
    if (sB < n) __builtin_nontemporal_store(o2, out + sB);
  }
}

extern "C" void kernel_launch(void* const* d_in, const int* in_sizes, int n_in,
                              void* d_out, int out_size, void* d_ws, size_t ws_size,
                              hipStream_t stream) {
  const float* x  = (const float*)d_in[0];   // [B,1,1,5] f32
  const float* wc = (const float*)d_in[1];   // [5,1,1,5] f32
  const float* wl = (const float*)d_in[2];   // [1,5]     f32
  float* out = (float*)d_out;                // [B,1]     f32

  const int n = in_sizes[0] / 5;             // B samples
  const int block = 256;                     // 8 waves, 256 samples/block
  const int grid  = (n + block - 1) / block;
  bnn_wmma_kernel<<<grid, block, 0, stream>>>(x, wc, wl, out, n);
}
